// MultiHeadAttention_18983755448537
// MI455X (gfx1250) — matrix operand order repaired and checked by reading, not compiled, not run on hardware
//
#include <hip/hip_runtime.h>
#include <hip/hip_bf16.h>

// ---------------------------------------------------------------------------
// MI455X (gfx1250, wave32) fused MHA:
//   qkv = [q|k|v] @ Wqkv^T + b  ->  flash attention  ->  out = attn @ Wo^T + b
// bf16 WMMA (v_wmma_f32_16x16x32_bf16) + async global->LDS staging (ASYNCcnt)
// with double-buffered LDS in the GEMMs.
// ---------------------------------------------------------------------------

typedef __attribute__((ext_vector_type(16))) __bf16 v16bf;
typedef __attribute__((ext_vector_type(8)))  float  v8f;
typedef __attribute__((ext_vector_type(4)))  int    v4i;

// Address-space-qualified v4i for the async-LDS builtin's exact signature:
//   (global v4i* src, local v4i* dst, imm offset, imm cpol)
typedef __attribute__((address_space(1))) v4i gv4i;
typedef __attribute__((address_space(3))) v4i lv4i;

#define B_     2
#define S_     2048
#define D_     1024
#define H_     16
#define HD_    64
#define NROWS  (B_ * S_)       // 4096
#define D3     (3 * D_)        // 3072

// ---------------------------------------------------------------------------
// Async global->LDS 16B copy (gfx1250 GLOBAL_LOAD_ASYNC_TO_LDS_B128, ASYNCcnt).
// Guarded: falls back to a synchronous vector copy if the builtin is absent.
// ---------------------------------------------------------------------------
#if __has_builtin(__builtin_amdgcn_global_load_async_to_lds_b128)
#define HAVE_ASYNC_LDS 1
#else
#define HAVE_ASYNC_LDS 0
#endif

__device__ __forceinline__ void async_copy16(void* dst_lds, const void* src_glob) {
#if HAVE_ASYNC_LDS
  __builtin_amdgcn_global_load_async_to_lds_b128(
      (gv4i*)src_glob, (lv4i*)dst_lds, /*offset=*/0, /*cpol=*/0);
#else
  *reinterpret_cast<uint4*>(dst_lds) = *reinterpret_cast<const uint4*>(src_glob);
#endif
}

__device__ __forceinline__ void wait_async_lds() {
#if HAVE_ASYNC_LDS
#if __has_builtin(__builtin_amdgcn_s_wait_asynccnt)
  __builtin_amdgcn_s_wait_asynccnt(0);
#else
  asm volatile("s_wait_asynccnt 0x0" ::: "memory");
#endif
#endif
}

// ---------------------------------------------------------------------------
// fp32 -> bf16 conversion (4-wide grid-stride)
// ---------------------------------------------------------------------------
struct alignas(8) bf4 { __bf16 v[4]; };

__global__ void cvt_f32_bf16_kernel(const float* __restrict__ src,
                                    __bf16* __restrict__ dst, int n4) {
  int i = blockIdx.x * blockDim.x + threadIdx.x;
  int stride = gridDim.x * blockDim.x;
  for (; i < n4; i += stride) {
    float4 f = reinterpret_cast<const float4*>(src)[i];
    bf4 o;
    o.v[0] = (__bf16)f.x; o.v[1] = (__bf16)f.y;
    o.v[2] = (__bf16)f.z; o.v[3] = (__bf16)f.w;
    reinterpret_cast<bf4*>(dst)[i] = o;
  }
}

// ---------------------------------------------------------------------------
// GEMM: out[n, j] = bias[j] + sum_seg sum_k A_seg[n,k] * W[j, seg*1024+k]
//   Block tile 128(M) x 64(N), 256 threads = 8 waves (4 M x 2 N),
//   wave tile 32x32 -> 4 accumulators, 4 WMMAs per K-step of 32.
//   Double-buffered LDS, async global->LDS staging overlapped with WMMA.
// ---------------------------------------------------------------------------
template <int NSEG, bool OUTF32>
__global__ __launch_bounds__(256)
void gemm_wt_kernel(const __bf16* __restrict__ A0, const __bf16* __restrict__ A1,
                    const __bf16* __restrict__ A2, const __bf16* __restrict__ W,
                    const float* __restrict__ bias,
                    __bf16* __restrict__ outb, float* __restrict__ outf) {
  constexpr int NCOLS = NSEG * 1024;
  __shared__ __bf16 Atile[2][128 * 32];   // 2 x 8 KB
  __shared__ __bf16 Wtile[2][64 * 32];    // 2 x 4 KB

  const int tid   = threadIdx.x;
  const int lane  = tid & 31;
  const int wid   = tid >> 5;
  const int wm    = wid & 3;          // wave M index (0..3) -> 32 rows each
  const int wn    = wid >> 2;         // wave N index (0..1) -> 32 cols each
  const int lanen = lane & 15;
  const int khalf = lane >> 4;
  const int m0 = blockIdx.y * 128;
  const int j0 = blockIdx.x * 64;

  const __bf16* As[3] = {A0, A1, A2};

  // Stage one 128x32 A tile + 64x32 W tile into buffer `buf` (async).
  auto stage = [&](int buf, int ks) {
    const int seg = ks >> 10;
    const int k0  = ks & 1023;
    const __bf16* A = As[seg];
#pragma unroll
    for (int i = 0; i < 2; ++i) {
      const int c   = tid + 256 * i;      // 512 chunks of 8 bf16
      const int row = c >> 2;
      const int cc  = c & 3;
      async_copy16(&Atile[buf][row * 32 + cc * 8],
                   A + (size_t)(m0 + row) * 1024 + k0 + cc * 8);
    }
    {
      const int row = tid >> 2;           // 256 chunks
      const int cc  = tid & 3;
      async_copy16(&Wtile[buf][row * 32 + cc * 8],
                   W + (size_t)(j0 + row) * NCOLS + seg * 1024 + k0 + cc * 8);
    }
  };

  v8f acc[2][2];
#pragma unroll
  for (int mt = 0; mt < 2; ++mt)
#pragma unroll
    for (int nt = 0; nt < 2; ++nt) acc[mt][nt] = {};

  stage(0, 0);
  int buf = 0;
  for (int ks = 0; ks < NSEG * 1024; ks += 32) {
    wait_async_lds();        // this wave's staged tile is in LDS
    __syncthreads();         // everyone's tile visible; prior reads retired
    if (ks + 32 < NSEG * 1024) stage(buf ^ 1, ks + 32);  // prefetch next tile

    // A fragments (16x32): lane = row, elems K = khalf*8+0..7, 16+khalf*8+0..7
    v16bf a[2];
#pragma unroll
    for (int mt = 0; mt < 2; ++mt) {
      const __bf16* p = &Atile[buf][(wm * 32 + mt * 16 + lanen) * 32 + khalf * 8];
      reinterpret_cast<uint4*>(&a[mt])[0] = *reinterpret_cast<const uint4*>(p);
      reinterpret_cast<uint4*>(&a[mt])[1] = *reinterpret_cast<const uint4*>(p + 16);
    }
    // B fragments (32x16): lane = column, elems K = khalf*8+0..7, 16+khalf*8+0..7
    v16bf bfr[2];
#pragma unroll
    for (int nt = 0; nt < 2; ++nt) {
      const __bf16* p = &Wtile[buf][(wn * 32 + nt * 16 + lanen) * 32 + khalf * 8];
      reinterpret_cast<uint4*>(&bfr[nt])[0] = *reinterpret_cast<const uint4*>(p);
      reinterpret_cast<uint4*>(&bfr[nt])[1] = *reinterpret_cast<const uint4*>(p + 16);
    }
#pragma unroll
    for (int mt = 0; mt < 2; ++mt)
#pragma unroll
      for (int nt = 0; nt < 2; ++nt)
        acc[mt][nt] = __builtin_amdgcn_wmma_f32_16x16x32_bf16(
            false, a[mt], false, bfr[nt], (short)0, acc[mt][nt], false, false);
    buf ^= 1;
  }

  // Epilogue: bias + store. C layout: VGPR r -> row M = r + 8*khalf, col = lanen.
#pragma unroll
  for (int nt = 0; nt < 2; ++nt) {
    const int jg = j0 + wn * 32 + nt * 16 + lanen;
    const float bv = bias[jg];
#pragma unroll
    for (int mt = 0; mt < 2; ++mt)
#pragma unroll
      for (int r = 0; r < 8; ++r) {
        const int m = m0 + wm * 32 + mt * 16 + r + 8 * khalf;
        if constexpr (OUTF32) {
          outf[(size_t)m * NCOLS + jg] = acc[mt][nt][r] + bv;
        } else {
          outb[(size_t)m * NCOLS + jg] = (__bf16)(acc[mt][nt][r] + bv);
        }
      }
  }
}

// ---------------------------------------------------------------------------
// Flash attention: grid (S/64, B*H), block 128 = 4 waves, wave = 16 query rows.
// qkvp: [NROWS, 3072] bf16 (q | k | v). attnws: [NROWS, 1024] bf16, heads merged.
// K tile staged via async global->LDS; V tile transposed manually.
// ---------------------------------------------------------------------------
__global__ __launch_bounds__(128)
void attn_kernel(const __bf16* __restrict__ qkvp, __bf16* __restrict__ attnws) {
  __shared__ __bf16 Klds[64 * 64];        // [key][hd]  (row-major)
  __shared__ __bf16 Vlds[64 * 64];        // [hd][key]  (transposed)
  __shared__ __bf16 Plds[4 * 16 * 64];    // per-wave prob patch [row][key]

  const int tid   = threadIdx.x;
  const int lane  = tid & 31;
  const int w     = tid >> 5;
  const int lanen = lane & 15;
  const int khalf = lane >> 4;
  const int b     = blockIdx.y >> 4;
  const int h     = blockIdx.y & 15;
  const int bbase = b * S_;
  const int q0w   = blockIdx.x * 64 + w * 16;

  // Q A-fragments: two 16x32 frags covering hd = 0..63 (loaded once).
  v16bf qa[2];
  {
    const size_t qrow = (size_t)(bbase + q0w + lanen) * D3 + h * HD_;
#pragma unroll
    for (int kf = 0; kf < 2; ++kf) {
      const __bf16* p = qkvp + qrow + kf * 32 + khalf * 8;
      reinterpret_cast<uint4*>(&qa[kf])[0] = *reinterpret_cast<const uint4*>(p);
      reinterpret_cast<uint4*>(&qa[kf])[1] = *reinterpret_cast<const uint4*>(p + 16);
    }
  }

  v8f oacc[4];
#pragma unroll
  for (int nt = 0; nt < 4; ++nt) oacc[nt] = {};
  float m_run[8], l_run[8];
#pragma unroll
  for (int r = 0; r < 8; ++r) { m_run[r] = -1e30f; l_run[r] = 0.0f; }

  for (int kb0 = 0; kb0 < S_; kb0 += 64) {
    __syncthreads();  // previous iteration's LDS reads retired before overwrite
    // Stage K (async, row-major) and V (manual transpose) 64x64 tiles.
#pragma unroll
    for (int i = 0; i < 4; ++i) {
      const int chunk = i * 128 + tid;          // 512 chunks of 8 bf16
      const int key = chunk >> 3;
      const int hd8 = chunk & 7;
      const size_t rowoff = (size_t)(bbase + kb0 + key) * D3 + h * HD_ + hd8 * 8;
      async_copy16(&Klds[key * 64 + hd8 * 8], qkvp + rowoff + D_);
      uint4 vv = *reinterpret_cast<const uint4*>(qkvp + rowoff + 2 * D_);
      const __bf16* ve = reinterpret_cast<const __bf16*>(&vv);
#pragma unroll
      for (int e = 0; e < 8; ++e) Vlds[(hd8 * 8 + e) * 64 + key] = ve[e];
    }
    // Prefetch next key block while this one is consumed (global_prefetch_b8).
    if (kb0 + 64 < S_) {
      const __bf16* pf = qkvp + (size_t)(bbase + kb0 + 64 + (tid >> 1)) * D3 +
                         D_ + h * HD_ + (tid & 1) * 32;
      __builtin_prefetch(pf, 0, 1);
    }
    wait_async_lds();
    __syncthreads();

    // Scores: 16 queries x 64 keys, K-dim = hd = 64 -> 2 WMMAs per 16-key tile.
    v8f sacc[4];
#pragma unroll
    for (int nt = 0; nt < 4; ++nt) {
      v8f s = {};
#pragma unroll
      for (int kf = 0; kf < 2; ++kf) {
        v16bf bk;  // B frag: column = key, K = hd khalf*8+0..7, 16+khalf*8+0..7
        const __bf16* p = &Klds[(nt * 16 + lanen) * 64 + kf * 32 + khalf * 8];
        reinterpret_cast<uint4*>(&bk)[0] = *reinterpret_cast<const uint4*>(p);
        reinterpret_cast<uint4*>(&bk)[1] = *reinterpret_cast<const uint4*>(p + 16);
        s = __builtin_amdgcn_wmma_f32_16x16x32_bf16(false, qa[kf], false, bk,
                                                    (short)0, s, false, false);
      }
      sacc[nt] = s * 0.125f;  // 1/sqrt(64)
    }

    // Online softmax. Row M = r + 8*khalf spans the 16 lanes of this half.
    float tmax[8];
#pragma unroll
    for (int r = 0; r < 8; ++r)
      tmax[r] = fmaxf(fmaxf(sacc[0][r], sacc[1][r]), fmaxf(sacc[2][r], sacc[3][r]));
#pragma unroll
    for (int off = 8; off >= 1; off >>= 1)
#pragma unroll
      for (int r = 0; r < 8; ++r)
        tmax[r] = fmaxf(tmax[r], __shfl_xor(tmax[r], off, 32));

    float scl[8], psum[8];
#pragma unroll
    for (int r = 0; r < 8; ++r) {
      const float mn = fmaxf(m_run[r], tmax[r]);
      scl[r] = __expf(m_run[r] - mn);
      m_run[r] = mn;
      psum[r] = 0.0f;
    }
#pragma unroll
    for (int nt = 0; nt < 4; ++nt)
#pragma unroll
      for (int r = 0; r < 8; ++r) {
        const float p = __expf(sacc[nt][r] - m_run[r]);
        psum[r] += p;
        // C-layout -> A-layout via wave-private LDS patch.
        Plds[w * 1024 + (r + 8 * khalf) * 64 + nt * 16 + lanen] = (__bf16)p;
      }
#pragma unroll
    for (int off = 8; off >= 1; off >>= 1)
#pragma unroll
      for (int r = 0; r < 8; ++r)
        psum[r] += __shfl_xor(psum[r], off, 32);
#pragma unroll
    for (int r = 0; r < 8; ++r) l_run[r] = l_run[r] * scl[r] + psum[r];
#pragma unroll
    for (int nt = 0; nt < 4; ++nt)
#pragma unroll
      for (int r = 0; r < 8; ++r) oacc[nt][r] *= scl[r];

    // O += P @ V : K-dim = key = 64 -> 2 A-frags x 4 hd tiles.
#pragma unroll
    for (int kf = 0; kf < 2; ++kf) {
      v16bf pa;
      const __bf16* pp = &Plds[w * 1024 + lanen * 64 + kf * 32 + khalf * 8];
      reinterpret_cast<uint4*>(&pa)[0] = *reinterpret_cast<const uint4*>(pp);
      reinterpret_cast<uint4*>(&pa)[1] = *reinterpret_cast<const uint4*>(pp + 16);
#pragma unroll
      for (int nt = 0; nt < 4; ++nt) {
        v16bf vb;  // B frag: column = hd, K = key khalf*8+0..7, 16+khalf*8+0..7
        const __bf16* vp = &Vlds[(nt * 16 + lanen) * 64 + kf * 32 + khalf * 8];
        reinterpret_cast<uint4*>(&vb)[0] = *reinterpret_cast<const uint4*>(vp);
        reinterpret_cast<uint4*>(&vb)[1] = *reinterpret_cast<const uint4*>(vp + 16);
        oacc[nt] = __builtin_amdgcn_wmma_f32_16x16x32_bf16(false, pa, false, vb,
                                                           (short)0, oacc[nt],
                                                           false, false);
      }
    }
  }

  // Normalize and store merged-head layout [row, h*64 + hd].
  float inv[8];
#pragma unroll
  for (int r = 0; r < 8; ++r) inv[r] = 1.0f / l_run[r];
#pragma unroll
  for (int nt = 0; nt < 4; ++nt)
#pragma unroll
    for (int r = 0; r < 8; ++r) {
      const int m = bbase + q0w + r + 8 * khalf;
      attnws[(size_t)m * D_ + h * HD_ + nt * 16 + lanen] =
          (__bf16)(oacc[nt][r] * inv[r]);
    }
}

// ---------------------------------------------------------------------------
// Launch
// ---------------------------------------------------------------------------
extern "C" void kernel_launch(void* const* d_in, const int* in_sizes, int n_in,
                              void* d_out, int out_size, void* d_ws, size_t ws_size,
                              hipStream_t stream) {
  (void)in_sizes; (void)n_in; (void)out_size; (void)ws_size;

  const float* q_f    = (const float*)d_in[0];
  const float* k_f    = (const float*)d_in[1];
  const float* v_f    = (const float*)d_in[2];
  const float* wqkv_f = (const float*)d_in[3];
  const float* bqkv   = (const float*)d_in[4];
  const float* wo_f   = (const float*)d_in[5];
  const float* bo     = (const float*)d_in[6];
  float* out = (float*)d_out;

  // Workspace layout (bf16 buffers, all 16B-aligned offsets).
  uint8_t* ws = (uint8_t*)d_ws;
  size_t off = 0;
  __bf16* qb   = (__bf16*)(ws + off); off += (size_t)NROWS * D_ * 2;  // 8 MB
  __bf16* kb   = (__bf16*)(ws + off); off += (size_t)NROWS * D_ * 2;  // 8 MB
  __bf16* vb   = (__bf16*)(ws + off); off += (size_t)NROWS * D_ * 2;  // 8 MB
  __bf16* wq   = (__bf16*)(ws + off); off += (size_t)D3 * D3 * 2;     // 18.9 MB
  __bf16* wo   = (__bf16*)(ws + off); off += (size_t)D_ * D_ * 2;     // 2 MB
  __bf16* qkvp = (__bf16*)(ws + off); off += (size_t)NROWS * D3 * 2;  // 25.2 MB
  __bf16* attn = (__bf16*)(ws + off); off += (size_t)NROWS * D_ * 2;  // 8 MB

  // 1) fp32 -> bf16 conversions.
  const int cvt_blocks = 1024, cvt_threads = 256;
  cvt_f32_bf16_kernel<<<cvt_blocks, cvt_threads, 0, stream>>>(q_f, qb, NROWS * D_ / 4);
  cvt_f32_bf16_kernel<<<cvt_blocks, cvt_threads, 0, stream>>>(k_f, kb, NROWS * D_ / 4);
  cvt_f32_bf16_kernel<<<cvt_blocks, cvt_threads, 0, stream>>>(v_f, vb, NROWS * D_ / 4);
  cvt_f32_bf16_kernel<<<cvt_blocks, cvt_threads, 0, stream>>>(wqkv_f, wq, D3 * D3 / 4);
  cvt_f32_bf16_kernel<<<cvt_blocks, cvt_threads, 0, stream>>>(wo_f, wo, D_ * D_ / 4);

  // 2) Fused QKV projection: [4096,3072] = concat(q,k,v) @ Wqkv^T + b.
  gemm_wt_kernel<3, false><<<dim3(D3 / 64, NROWS / 128), 256, 0, stream>>>(
      qb, kb, vb, wq, bqkv, qkvp, nullptr);

  // 3) Flash attention over (b,h) x 64-query blocks.
  attn_kernel<<<dim3(S_ / 64, B_ * H_), 128, 0, stream>>>(qkvp, attn);

  // 4) Output projection: [4096,1024] = attn @ Wo^T + b (fp32 out).
  gemm_wt_kernel<1, true><<<dim3(D_ / 64, NROWS / 128), 256, 0, stream>>>(
      attn, nullptr, nullptr, wo, bo, nullptr, out);
}
